// MultiheadAttention_55559696941166
// MI455X (gfx1250) — compile-verified, matrix operand order checked by reading, not run on hardware
//
#include <hip/hip_runtime.h>
#include <hip/hip_bf16.h>
#include <math.h>

// Problem constants (from reference)
#define BATCH  16
#define SEQ    4096
#define HDIM   1024
#define HEADS  16
#define DH     64          // head dim
#define SSEG   4           // seq segments in K3 (parallelism over K)
#define SEGLEN (SEQ / SSEG)

typedef __attribute__((ext_vector_type(16))) __bf16 v16bf;
typedef __attribute__((ext_vector_type(8)))  __bf16 v8bf;
typedef __attribute__((ext_vector_type(8)))  float  v8f;
typedef __attribute__((ext_vector_type(2)))  float  v2f;

// ---------------------------------------------------------------------------
// K1: q[b] = query[b] @ Wq^T + bq   (per (b,h) block)
//     u[b,h,c]  = sum_d q[b,h*64+d] * Wk[h*64+d, c]      (stored bf16)
//     qbk[b,h]  = sum_d q[b,h*64+d] * bk[h*64+d]         (score bias)
// ---------------------------------------------------------------------------
__global__ void k1_proj_qu(const float* __restrict__ query,
                           const float* __restrict__ Wq,
                           const float* __restrict__ bq,
                           const float* __restrict__ Wk,
                           const float* __restrict__ bk,
                           __bf16* __restrict__ u_bf,
                           float*  __restrict__ qbk)
{
    const int b   = blockIdx.x >> 4;
    const int h   = blockIdx.x & 15;
    const int tid = threadIdx.x;

    __shared__ float s_q[DH];

    if (tid < DH) {
        const float* wr = Wq + (size_t)(h * DH + tid) * HDIM;
        const float* qp = query + (size_t)b * HDIM;
        float acc = 0.f;
        for (int e = 0; e < HDIM; ++e) acc += wr[e] * qp[e];
        s_q[tid] = acc + bq[h * DH + tid];
    }
    __syncthreads();

    if (tid == 0) {
        float acc = 0.f;
        for (int d = 0; d < DH; ++d) acc += s_q[d] * bk[h * DH + d];
        qbk[b * HEADS + h] = acc;
    }

    for (int c = tid; c < HDIM; c += blockDim.x) {
        float acc = 0.f;
        #pragma unroll 8
        for (int d = 0; d < DH; ++d)
            acc += s_q[d] * Wk[(size_t)(h * DH + d) * HDIM + c];
        u_bf[(size_t)(b * HEADS + h) * HDIM + c] = (__bf16)acc;
    }
}

// ---------------------------------------------------------------------------
// K2: scores tile (16 seq rows x 16 heads) = key_tile x u^T via
//     v_wmma_f32_16x16x32_bf16, then softmax over heads (cross-lane),
//     store att transposed: attT[b][h][s]  (f32).
//     One wave per 16-row tile; 8 waves per block.
// ---------------------------------------------------------------------------
__global__ void k2_scores_softmax(const float*  __restrict__ key,
                                  const __bf16* __restrict__ u_bf,
                                  const float*  __restrict__ qbk,
                                  float*        __restrict__ attT)
{
    const int wave = threadIdx.x >> 5;
    const int lane = threadIdx.x & 31;
    const int half = lane >> 4;           // K-half selector (ISA A/B layout)
    const int lm   = lane & 15;           // A row (seq) / B col (head)
    const int kb   = half * 8;

    const int tile = blockIdx.x * 8 + wave;   // 4096 tiles = B * S/16
    const int b    = tile >> 8;
    const int s0   = (tile & 255) * 16;

    const float*  Arow = key  + ((size_t)b * SEQ + s0 + lm) * HDIM;
    const __bf16* Brow = u_bf + (size_t)(b * HEADS + lm) * HDIM;

    v8f acc = {};
    for (int kc = 0; kc < HDIM; kc += 32) {
        // prefetch the key_ stream ~8 iterations (1 KB) ahead
        if (kc + 256 < HDIM)
            __builtin_prefetch(Arow + kc + 256, 0, 1);

        // A fragment: 16x32 bf16, lane L -> row L%16, K = kb+{0..7}, 16+kb+{0..7}
        v16bf a;
        {
            const float4* p0 = reinterpret_cast<const float4*>(Arow + kc + kb);
            const float4* p1 = reinterpret_cast<const float4*>(Arow + kc + 16 + kb);
            float4 f0 = p0[0], f1 = p0[1], f2 = p1[0], f3 = p1[1];
            a[0]  = (__bf16)f0.x; a[1]  = (__bf16)f0.y; a[2]  = (__bf16)f0.z; a[3]  = (__bf16)f0.w;
            a[4]  = (__bf16)f1.x; a[5]  = (__bf16)f1.y; a[6]  = (__bf16)f1.z; a[7]  = (__bf16)f1.w;
            a[8]  = (__bf16)f2.x; a[9]  = (__bf16)f2.y; a[10] = (__bf16)f2.z; a[11] = (__bf16)f2.w;
            a[12] = (__bf16)f3.x; a[13] = (__bf16)f3.y; a[14] = (__bf16)f3.z; a[15] = (__bf16)f3.w;
        }
        // B fragment: 32x16 bf16, lane L -> col L%16, same K pattern; u stored bf16
        v16bf bb;
        {
            v8bf b0 = *reinterpret_cast<const v8bf*>(Brow + kc + kb);
            v8bf b1 = *reinterpret_cast<const v8bf*>(Brow + kc + 16 + kb);
            #pragma unroll
            for (int i = 0; i < 8; ++i) { bb[i] = b0[i]; bb[8 + i] = b1[i]; }
        }
        acc = __builtin_amdgcn_wmma_f32_16x16x32_bf16(false, a, false, bb,
                                                      (short)0, acc, false, false);
    }

    // C layout: acc[r] -> M = r + half*8 (seq row), N = lm (head)
    const float qb = qbk[b * HEADS + lm];
    float e[8];
    #pragma unroll
    for (int r = 0; r < 8; ++r) {
        float sc = (acc[r] + qb) * 0.03125f;   // 1/sqrt(1024)
        // softmax over the 16 heads of this row: 16-lane xor reduction
        float m = sc;
        #pragma unroll
        for (int o = 8; o >= 1; o >>= 1) m = fmaxf(m, __shfl_xor(m, o, 32));
        float ex = __expf(sc - m);
        float su = ex;
        #pragma unroll
        for (int o = 8; o >= 1; o >>= 1) su += __shfl_xor(su, o, 32);
        e[r] = ex / su;
    }

    // store attT[b][head=lm][s0 + half*8 + r]  (contiguous 8 floats per lane)
    float* dst = attT + (size_t)(b * HEADS + lm) * SEQ + s0 + half * 8;
    *reinterpret_cast<float4*>(dst)     = float4{e[0], e[1], e[2], e[3]};
    *reinterpret_cast<float4*>(dst + 4) = float4{e[4], e[5], e[6], e[7]};
}

// ---------------------------------------------------------------------------
// K3: wpart[seg][b][h][c] = sum_{s in seg} attT[b][h][s] * value[b][s][c]
//     via v_wmma_f32_16x16x4_f32 (M=16 heads, N=16 cols, K=seq chunk of 4).
//     One wave per (seg, b, 16-col tile) -> 4096 waves streaming `value` once.
// ---------------------------------------------------------------------------
__global__ void k3_wsum(const float* __restrict__ attT,
                        const float* __restrict__ value,
                        float*       __restrict__ wbuf)
{
    const int wave = threadIdx.x >> 5;
    const int lane = threadIdx.x & 31;
    const int half = lane >> 4;
    const int lm   = lane & 15;

    const int tile = blockIdx.x * 8 + wave;   // SSEG * B * 64 = 4096 tiles
    const int seg  = tile >> 10;
    const int bc   = tile & 1023;
    const int b    = bc >> 6;
    const int c0   = (bc & 63) * 16;

    const float* Aptr = attT  + (size_t)(b * HEADS + lm) * SEQ + seg * SEGLEN;
    const float* Bptr = value + (size_t)b * SEQ * HDIM
                              + (size_t)seg * SEGLEN * HDIM + c0 + lm;

    v8f acc = {};
    #pragma unroll 4
    for (int s = 0; s < SEGLEN; s += 4) {
        const int k0 = s + half * 2;
        // prefetch the value stream ~8 iterations (32 rows) ahead
        if (s + 32 < SEGLEN)
            __builtin_prefetch(Bptr + (size_t)(k0 + 32) * HDIM, 0, 1);

        float2 af = *reinterpret_cast<const float2*>(Aptr + k0);
        v2f a; a[0] = af.x; a[1] = af.y;
        const float* bp = Bptr + (size_t)k0 * HDIM;
        v2f bv; bv[0] = bp[0]; bv[1] = bp[HDIM];
        acc = __builtin_amdgcn_wmma_f32_16x16x4_f32(false, a, false, bv,
                                                    (short)0, acc, false, false);
    }

    // D layout: acc[r] -> M = r + half*8 (head), N = lm (col)
    float* wp = wbuf + ((size_t)seg * BATCH + b) * HEADS * HDIM + c0 + lm;
    #pragma unroll
    for (int r = 0; r < 8; ++r)
        wp[(size_t)(r + half * 8) * HDIM] = acc[r];
}

// ---------------------------------------------------------------------------
// K3R: reduce the SSEG partial w buffers -> wsum[b][h][c]
// ---------------------------------------------------------------------------
__global__ void k3r_reduce(const float* __restrict__ wbuf,
                           float*       __restrict__ wsum)
{
    const size_t idx    = (size_t)blockIdx.x * blockDim.x + threadIdx.x;
    const size_t stride = (size_t)BATCH * HEADS * HDIM;   // per-segment stride
    float s = 0.f;
    #pragma unroll
    for (int seg = 0; seg < SSEG; ++seg) s += wbuf[idx + seg * stride];
    wsum[idx] = s;
}

// ---------------------------------------------------------------------------
// K4: sumatt[b,h] = sum_s attT;  o[j] = Wv[j,:].wsum[b,j/64,:] + sumatt*bv[j];
//     out[b,:] = o @ Wf^T + bf.   One block per batch.
// ---------------------------------------------------------------------------
__global__ void k4_final(const float* __restrict__ attT,
                         const float* __restrict__ wsum,
                         const float* __restrict__ Wv,
                         const float* __restrict__ bv,
                         const float* __restrict__ Wf,
                         const float* __restrict__ bf,
                         float*       __restrict__ out)
{
    const int b   = blockIdx.x;
    const int tid = threadIdx.x;

    __shared__ float red[256];
    __shared__ float s_sum[HEADS];
    __shared__ float s_o[HDIM];

    {   // sumatt
        const int h = tid & 15, part = tid >> 4;
        const float* ap = attT + (size_t)(b * HEADS + h) * SEQ + part * 256;
        float acc = 0.f;
        for (int i = 0; i < 256; ++i) acc += ap[i];
        red[tid] = acc;
    }
    __syncthreads();
    if (tid < HEADS) {
        float s = 0.f;
        for (int p = 0; p < 16; ++p) s += red[p * 16 + tid];
        s_sum[tid] = s;
    }
    __syncthreads();

    for (int j = tid; j < HDIM; j += blockDim.x) {
        const int h = j >> 6;
        const float* wr = Wv   + (size_t)j * HDIM;
        const float* wv = wsum + (size_t)(b * HEADS + h) * HDIM;
        float acc = 0.f;
        #pragma unroll 4
        for (int c = 0; c < HDIM; ++c) acc += wr[c] * wv[c];
        s_o[j] = acc + s_sum[h] * bv[j];
    }
    __syncthreads();

    for (int j = tid; j < HDIM; j += blockDim.x) {
        const float* fr = Wf + (size_t)j * HDIM;
        float acc = 0.f;
        #pragma unroll 4
        for (int i = 0; i < HDIM; ++i) acc += fr[i] * s_o[i];
        out[(size_t)b * HDIM + j] = acc + bf[j];
    }
}

// ---------------------------------------------------------------------------
extern "C" void kernel_launch(void* const* d_in, const int* in_sizes, int n_in,
                              void* d_out, int out_size, void* d_ws, size_t ws_size,
                              hipStream_t stream)
{
    (void)in_sizes; (void)n_in; (void)out_size; (void)ws_size;

    const float* query = (const float*)d_in[0];
    const float* key_  = (const float*)d_in[1];
    const float* value = (const float*)d_in[2];
    const float* Wq    = (const float*)d_in[3];
    const float* bq    = (const float*)d_in[4];
    const float* Wk    = (const float*)d_in[5];
    const float* bk    = (const float*)d_in[6];
    const float* Wv    = (const float*)d_in[7];
    const float* bv    = (const float*)d_in[8];
    const float* Wf    = (const float*)d_in[9];
    const float* bf    = (const float*)d_in[10];
    float* out = (float*)d_out;

    // Workspace layout (10 MB total):
    //   [0, 512K)        u_bf   bf16 [B][H][HDIM]
    //   [512K, 512K+1K)  qbk    f32  [B][H]
    //   [1M, 5M)         attT   f32  [B][H][SEQ]
    //   [5M, 9M)         wbuf   f32  [SSEG][B][H][HDIM]
    //   [9M, 10M)        wsum   f32  [B][H][HDIM]
    char* ws = (char*)d_ws;
    __bf16* u_bf = (__bf16*)(ws);
    float*  qbk  = (float*)(ws + (512u << 10));
    float*  attT = (float*)(ws + (1u << 20));
    float*  wbuf = (float*)(ws + (5u << 20));
    float*  wsum = (float*)(ws + (9u << 20));

    k1_proj_qu       <<<dim3(BATCH * HEADS), dim3(256), 0, stream>>>(query, Wq, bq, Wk, bk, u_bf, qbk);
    k2_scores_softmax<<<dim3(512),           dim3(256), 0, stream>>>(key_, u_bf, qbk, attT);
    k3_wsum          <<<dim3(512),           dim3(256), 0, stream>>>(attT, value, wbuf);
    k3r_reduce       <<<dim3(1024),          dim3(256), 0, stream>>>(wbuf, wsum);
    k4_final         <<<dim3(BATCH),         dim3(256), 0, stream>>>(attT, wsum, Wv, bv, Wf, bf, out);
}
